// EEMoE_90512140795914
// MI455X (gfx1250) — matrix operand order repaired and checked by reading, not compiled, not run on hardware
//
#include <hip/hip_runtime.h>
#include <cstdint>
#include <cstddef>

// MoE: B=16,H=64,W=64,C=384,E=5,TOPK=2  ->  N=65536 tokens
#define N_TOK   65536
#define C_DIM   384
#define E_NUM   5
#define TOK_BLK 64    // tokens per workgroup = 4 M-subtiles of 16
#define MT_N    4     // M-subtiles per workgroup (B-fragment reuse factor)
#define A_STR   392   // padded LDS row stride (halfs): 784B rows -> banks rotate by 4/row

typedef _Float16 v16h __attribute__((ext_vector_type(16)));
typedef _Float16 h8   __attribute__((ext_vector_type(8)));
typedef _Float16 h4   __attribute__((ext_vector_type(4)));
typedef float    v8f  __attribute__((ext_vector_type(8)));

union AB { v16h v; h8 p[2]; };

// We[e][c][d] (f32) -> Wt[e][d][c] (f16), so B fragments are contiguous along K=c.
__global__ __launch_bounds__(256) void moe_wt_convert(const float* __restrict__ We,
                                                      _Float16* __restrict__ Wt) {
    int o = blockIdx.x * blockDim.x + threadIdx.x;
    if (o >= E_NUM * C_DIM * C_DIM) return;
    int c = o % C_DIM;
    int d = (o / C_DIM) % C_DIM;
    int e = o / (C_DIM * C_DIM);
    Wt[o] = (_Float16)We[(size_t)e * C_DIM * C_DIM + (size_t)c * C_DIM + d];
}

__global__ __launch_bounds__(256, 2) void moe_main(const float* __restrict__ x,
                                                   const float* __restrict__ Wg,
                                                   const _Float16* __restrict__ Wt,
                                                   const float* __restrict__ be,
                                                   float* __restrict__ out) {
    __shared__ _Float16 sA[TOK_BLK * A_STR];      // 64 tokens x 384 ch, f16, padded
    __shared__ float    sLog[TOK_BLK * E_NUM];    // router logits
    __shared__ float    sComb[TOK_BLK * E_NUM];   // sparse combine weights (gates)

    const int tid  = threadIdx.x;
    const int tok0 = blockIdx.x * TOK_BLK;

    // ---- stage x block (64x384 f32) into LDS as f16, float4-vectorized ----
    {
        const float4* xv = (const float4*)(x + (size_t)tok0 * C_DIM);
        #pragma unroll
        for (int i = 0; i < (TOK_BLK * C_DIM / 4) / 256; ++i) {   // 24 iters
            int idx4 = tid + 256 * i;
            int m  = idx4 / (C_DIM / 4);
            int c4 = idx4 % (C_DIM / 4);
            float4 v = xv[idx4];
            h4 hv = { (_Float16)v.x, (_Float16)v.y, (_Float16)v.z, (_Float16)v.w };
            *(h4*)&sA[m * A_STR + 4 * c4] = hv;
        }
    }

    // ---- routing logits: 320 (token, expert) dot products over all 256 threads,
    //      f32 from f32 x (matches reference top-k selection closely) ----
    #pragma unroll
    for (int half = 0; half < 2; ++half) {
        int p = tid + 256 * half;
        if (p < TOK_BLK * E_NUM) {
            int m = p / E_NUM;
            int e = p % E_NUM;
            const float* xr = x + (size_t)(tok0 + m) * C_DIM;
            float acc = 0.f;
            for (int c = 0; c < C_DIM; ++c)
                acc = fmaf(xr[c], Wg[c * E_NUM + e], acc);
            sLog[p] = acc;
        }
    }
    __syncthreads();

    // ---- top-2 + softmax per token ----
    if (tid < TOK_BLK) {
        float lg[E_NUM];
        #pragma unroll
        for (int e = 0; e < E_NUM; ++e) lg[e] = sLog[tid * E_NUM + e];
        float v0 = lg[0]; int i0 = 0;
        float v1 = -3.4e38f; int i1 = 1;
        #pragma unroll
        for (int e = 1; e < E_NUM; ++e) {
            if (lg[e] > v0)      { v1 = v0; i1 = i0; v0 = lg[e]; i0 = e; }
            else if (lg[e] > v1) { v1 = lg[e]; i1 = e; }
        }
        float e1x = __expf(v1 - v0);
        float g0  = 1.0f / (1.0f + e1x);
        float g1  = e1x * g0;
        #pragma unroll
        for (int e = 0; e < E_NUM; ++e) sComb[tid * E_NUM + e] = 0.f;
        sComb[tid * E_NUM + i0] = g0;
        sComb[tid * E_NUM + i1] = g1;
    }
    __syncthreads();

    // ---- WMMA compute: each wave owns 3 of the 24 output-channel tiles.
    //      tile -> expert -> K; one B fragment reused across 4 M-subtiles.
    //      ks/e loops kept ROLLED so A fragments are not hoisted (register blowup). ----
    const int wave = tid >> 5;       // wave32
    const int lane = tid & 31;
    const int hh   = lane >> 4;      // half-wave select
    const int nn   = lane & 15;      // N column (B/C/D layout)
    const int M    = lane & 15;      // A row within a 16-token subtile

    // per-lane A row base pointers (K chunk offsets: +0 and +16 from here)
    const _Float16* arowB[MT_N];
    #pragma unroll
    for (int mt = 0; mt < MT_N; ++mt)
        arowB[mt] = &sA[(mt * 16 + M) * A_STR + 8 * hh];

    for (int t = 0; t < 3; ++t) {
        const int oc0 = (wave + 8 * t) * 16;

        v8f yacc[MT_N];                       // running f32 output (gates applied)
        #pragma unroll
        for (int mt = 0; mt < MT_N; ++mt) yacc[mt] = v8f{};

        #pragma unroll 1
        for (int e = 0; e < E_NUM; ++e) {
            v8f acc[MT_N];
            #pragma unroll
            for (int mt = 0; mt < MT_N; ++mt) acc[mt] = v8f{};

            // B column: lane holds column oc0+nn, K = kb+8hh..+7 and kb+16+8hh..+7
            const _Float16* bcol =
                Wt + ((size_t)(e * C_DIM + oc0 + nn)) * C_DIM + 8 * hh;

            #pragma unroll 1
            for (int ks = 0; ks < C_DIM / 32; ++ks) {
                const int kb = ks * 32;
                AB b;
                b.p[0] = *(const h8*)(bcol + kb);
                b.p[1] = *(const h8*)(bcol + kb + 16);
                #pragma unroll
                for (int mt = 0; mt < MT_N; ++mt) {
                    AB a;
                    a.p[0] = *(const h8*)(arowB[mt] + kb);        // K = kb+8hh..+7
                    a.p[1] = *(const h8*)(arowB[mt] + kb + 16);   // K = kb+16+8hh..+7
                    acc[mt] = __builtin_amdgcn_wmma_f32_16x16x32_f16(
                        false, a.v, false, b.v, (short)0, acc[mt], false, false);
                }
            }

            // fold this expert into the running output with its per-token gate
            const float beV = be[e * C_DIM + oc0 + nn];
            #pragma unroll
            for (int mt = 0; mt < MT_N; ++mt) {
                #pragma unroll
                for (int v = 0; v < 8; ++v) {
                    const int m = mt * 16 + v + 8 * hh;   // C/D layout row
                    yacc[mt][v] += sComb[m * E_NUM + e] * (acc[mt][v] + beV);
                }
            }
        }

        // ---- epilogue: LeakyReLU, store f32 ----
        #pragma unroll
        for (int mt = 0; mt < MT_N; ++mt) {
            #pragma unroll
            for (int v = 0; v < 8; ++v) {
                const int m = mt * 16 + v + 8 * hh;
                float y = yacc[mt][v];
                y = (y > 0.f) ? y : 0.01f * y;
                out[(size_t)(tok0 + m) * C_DIM + oc0 + nn] = y;
            }
        }
    }
}

extern "C" void kernel_launch(void* const* d_in, const int* in_sizes, int n_in,
                              void* d_out, int out_size, void* d_ws, size_t ws_size,
                              hipStream_t stream) {
    const float* x  = (const float*)d_in[0];   // [N, C] tokens (B*H*W flattened)
    const float* Wg = (const float*)d_in[1];   // [C, E]
    const float* We = (const float*)d_in[2];   // [E, C, C]
    const float* be = (const float*)d_in[3];   // [E, C]
    float* out = (float*)d_out;                // [N, C]

    _Float16* Wt = (_Float16*)d_ws;            // [E, C, C] f16, transposed (1.47 MB)

    const int wt_elems = E_NUM * C_DIM * C_DIM;
    moe_wt_convert<<<(wt_elems + 255) / 256, 256, 0, stream>>>(We, Wt);
    moe_main<<<N_TOK / TOK_BLK, 256, 0, stream>>>(x, Wg, Wt, be, out);
}
